// G2ANet_Critic_83167746719907
// MI455X (gfx1250) — matrix operand order repaired and checked by reading, not compiled, not run on hardware
//
#include <hip/hip_runtime.h>
#include <hip/hip_fp16.h>

// ---------------------------------------------------------------------------
// G2ANet critic, fused for MI455X (gfx1250, wave32, WMMA f16->f32).
// B=128 N=32 H=256 OBS=512, rows = 4096, T = N-1 = 31.
// ---------------------------------------------------------------------------

typedef _Float16 half_t;
typedef __attribute__((ext_vector_type(16))) _Float16 v16h;
typedef __attribute__((ext_vector_type(2)))  _Float16 h2v;
typedef __attribute__((ext_vector_type(8)))  float    v8f;

#define ROWS 4096
#define HD   256
#define OBSD 512
#define TT   31

static __device__ __forceinline__ float sigf(float x) {
    return 1.0f / (1.0f + __expf(-x));
}

// ---- WMMA helpers ---------------------------------------------------------
// A fragment: 16x32 f16 tile from LDS (row-major, leading dim ld halves).
// Per ISA: lanes 0-15 -> M=0..15; VGPR r holds K = 2r (+8 if r>=4) (+8 if lane>=16).
static __device__ __forceinline__ v16h frag_a_lds(const half_t* s, int ld, int k0, int lane) {
    int m = lane & 15, hi = lane >> 4;
    const half_t* base = s + m * ld + k0 + hi * 8;
    v16h a;
#pragma unroll
    for (int r = 0; r < 8; ++r) {
        int kb = 2 * r + (r >= 4 ? 8 : 0);
        h2v u = *(const h2v*)(base + kb);
        a[2 * r] = u.x; a[2 * r + 1] = u.y;
    }
    return a;
}

// B fragment: 32x16 (K x N), B[k][n] = W[n][k], W row-major with stride wstride,
// column offset woff. Same K order as A: VGPR r holds K = 2r (+8 if r>=4) (+8 if lane>=16); n = lane&15.
static __device__ __forceinline__ v16h frag_b_w(const half_t* W, int wstride, int woff,
                                                int n0, int k0, int lane) {
    int n = lane & 15, hi = lane >> 4;
    const half_t* base = W + (size_t)(n0 + n) * wstride + woff + k0 + hi * 8;
    v16h b;
#pragma unroll
    for (int r = 0; r < 8; ++r) {
        h2v u = *(const h2v*)(base + 2 * r + (r >= 4 ? 8 : 0));
        b[2 * r] = u.x; b[2 * r + 1] = u.y;
    }
    return b;
}

static __device__ __forceinline__ v8f wmma_f16(v16h a, v16h b, v8f c) {
    return __builtin_amdgcn_wmma_f32_16x16x32_f16(false, a, false, b, (short)0, c,
                                                  false, false);
}

// ---- f32 -> f16 weight conversion ----------------------------------------
__global__ void cvt_f16_kernel(const float* __restrict__ src, half_t* __restrict__ dst, int n) {
    int i = blockIdx.x * blockDim.x + threadIdx.x;
    if (i < n) dst[i] = (half_t)src[i];
}

// ---- Encoder + LSTM cell (16 rows per block) ------------------------------
__global__ __launch_bounds__(256)
void enc_lstm_kernel(const float* __restrict__ obs, const float* __restrict__ lstm_h,
                     const float* __restrict__ lstm_c, const float* __restrict__ masks,
                     const half_t* __restrict__ We16, const float* __restrict__ b_enc,
                     const half_t* __restrict__ Wi16, const half_t* __restrict__ Wh16,
                     const float* __restrict__ bi, const float* __restrict__ bh,
                     float* __restrict__ out_h, float* __restrict__ out_c,
                     half_t* __restrict__ hb16) {
    __shared__ half_t s_obs[16 * OBSD];   // 16 KB
    __shared__ half_t s_x[16 * HD];       // 8 KB
    __shared__ half_t s_h0[16 * HD];      // 8 KB
    __shared__ float  s_g[16 * 1024];     // 64 KB
    int tid = threadIdx.x, lane = tid & 31, wave = tid >> 5;
    int row0 = blockIdx.x * 16;

    for (int i = tid; i < 16 * OBSD; i += 256)
        s_obs[i] = (half_t)obs[(size_t)row0 * OBSD + i];
    for (int i = tid; i < 16 * HD; i += 256) {
        int r = i >> 8;
        s_h0[i] = (half_t)(lstm_h[(size_t)row0 * HD + i] * masks[row0 + r]);
    }
    __syncthreads();

    // x = relu(obs @ We^T + b_enc)  -> s_x (f16). 16 col-tiles, 2 per wave.
#pragma unroll
    for (int t = 0; t < 2; ++t) {
        int n0 = (wave * 2 + t) * 16;
        v8f acc = {};
        for (int k0 = 0; k0 < OBSD; k0 += 32) {
            v16h a = frag_a_lds(s_obs, OBSD, k0, lane);
            v16h b = frag_b_w(We16, OBSD, 0, n0, k0, lane);
            acc = wmma_f16(a, b, acc);
        }
        int col = (lane & 15) + n0, hi = lane >> 4;
        float be = b_enc[col];
#pragma unroll
        for (int r = 0; r < 8; ++r) {
            float v = acc[r] + be;
            s_x[(r + hi * 8) * HD + col] = (half_t)(v > 0.f ? v : 0.f);
        }
    }
    __syncthreads();

    // gates = x @ Wi^T + h0 @ Wh^T + bi + bh -> s_g. 64 tiles, 8 per wave.
    for (int t = 0; t < 8; ++t) {
        int n0 = (wave * 8 + t) * 16;
        v8f acc = {};
        for (int k0 = 0; k0 < HD; k0 += 32) {
            acc = wmma_f16(frag_a_lds(s_x, HD, k0, lane),
                           frag_b_w(Wi16, HD, 0, n0, k0, lane), acc);
            acc = wmma_f16(frag_a_lds(s_h0, HD, k0, lane),
                           frag_b_w(Wh16, HD, 0, n0, k0, lane), acc);
        }
        int col = (lane & 15) + n0, hi = lane >> 4;
        float bsum = bi[col] + bh[col];
#pragma unroll
        for (int r = 0; r < 8; ++r) s_g[(r + hi * 8) * 1024 + col] = acc[r] + bsum;
    }
    __syncthreads();

    // LSTM gate math
    for (int i = tid; i < 16 * HD; i += 256) {
        int r = i >> 8, c = i & 255;
        size_t grow = (size_t)row0 + r;
        float ig = s_g[r * 1024 + c],       fg = s_g[r * 1024 + 256 + c];
        float gg = s_g[r * 1024 + 512 + c], og = s_g[r * 1024 + 768 + c];
        float c0 = lstm_c[grow * HD + c] * masks[grow];
        float cc = sigf(fg) * c0 + sigf(ig) * tanhf(gg);
        float hh = sigf(og) * tanhf(cc);
        out_c[grow * HD + c] = cc;
        out_h[grow * HD + c] = hh;
        hb16[grow * HD + c]  = (half_t)hh;
    }
}

// ---- Generic WMMA GEMM: C[ROWS x Nout] = A16[ROWS x 256] @ W^T (+bias) ----
__global__ __launch_bounds__(256)
void gemm16_kernel(const half_t* __restrict__ A16, const half_t* __restrict__ W16,
                   const float* __restrict__ bias, float* __restrict__ C,
                   int wstride, int woff, int Nout) {
    __shared__ half_t sA[16 * HD];
    int tid = threadIdx.x, lane = tid & 31, wave = tid >> 5;
    int row0 = blockIdx.x * 16;
    for (int i = tid * 2; i < 16 * HD; i += 512)
        *(h2v*)(sA + i) = *(const h2v*)(A16 + (size_t)row0 * HD + i);
    __syncthreads();

    v16h afr[8];
#pragma unroll
    for (int kc = 0; kc < 8; ++kc) afr[kc] = frag_a_lds(sA, HD, kc * 32, lane);

#pragma unroll
    for (int t = 0; t < 2; ++t) {
        int n0 = blockIdx.y * 256 + (wave * 2 + t) * 16;
        if (n0 >= Nout) break;
        v8f acc = {};
#pragma unroll
        for (int kc = 0; kc < 8; ++kc)
            acc = wmma_f16(afr[kc], frag_b_w(W16, wstride, woff, n0, kc * 32, lane), acc);
        int col = (lane & 15) + n0, hi = lane >> 4;
        float bb = bias ? bias[col] : 0.f;
#pragma unroll
        for (int r = 0; r < 8; ++r)
            C[(size_t)(row0 + r + hi * 8) * Nout + col] = acc[r] + bb;
    }
}

// ---- GRU direction: 16 rows per block, sequential over T=31 steps ---------
// xg[t,row] = P[row] + Q[b*32 + nonself(n,t)]  (biases folded into P)
// hg        = h @ Wh^T + bh   (WMMA, hidden state resident in LDS)
// Also emits per-step logit partials lout[row][t][o] = h_t . W_hard[o, off:off+256]
__global__ __launch_bounds__(256)
void gru_kernel(const float* __restrict__ P, const float* __restrict__ Q,
                const half_t* __restrict__ Wh16, const float* __restrict__ bh,
                const float* __restrict__ W_hard, int hard_off,
                float* __restrict__ lout, int reverse) {
    __shared__ half_t s_h16[16 * HD];   // 8 KB
    __shared__ float  s_h32[16 * HD];   // 16 KB
    __shared__ float  s_hg[16 * 768];   // 48 KB
    int tid = threadIdx.x, lane = tid & 31, wave = tid >> 5;
    int row0 = blockIdx.x * 16;

    for (int i = tid; i < 16 * HD; i += 256) { s_h16[i] = (half_t)0.f; s_h32[i] = 0.f; }
    __syncthreads();

    for (int step = 0; step < TT; ++step) {
        int t = reverse ? (TT - 1 - step) : step;

        // hg = h @ Wh^T + bh : 48 tiles of 16x16, 6 per wave; A-frags hoisted.
        v16h afr[8];
#pragma unroll
        for (int kc = 0; kc < 8; ++kc) afr[kc] = frag_a_lds(s_h16, HD, kc * 32, lane);
        for (int tt = 0; tt < 6; ++tt) {
            int n0 = (wave * 6 + tt) * 16;
            v8f acc = {};
#pragma unroll
            for (int kc = 0; kc < 8; ++kc)
                acc = wmma_f16(afr[kc], frag_b_w(Wh16, HD, 0, n0, kc * 32, lane), acc);
            int col = (lane & 15) + n0, hi = lane >> 4;
            float bb = bh[col];
#pragma unroll
            for (int r = 0; r < 8; ++r) s_hg[(r + hi * 8) * 768 + col] = acc[r] + bb;
        }
        __syncthreads();

        // GRU cell elementwise
        for (int i = tid; i < 16 * HD; i += 256) {
            int r = i >> 8, c = i & 255;
            int grow = row0 + r;
            int bidx = grow >> 5, nidx = grow & 31;
            int j = (t < nidx) ? t : t + 1;
            size_t prow = (size_t)grow * 768;
            size_t qrow = (size_t)(bidx * 32 + j) * 768;
            float xr = P[prow + c]       + Q[qrow + c];
            float xz = P[prow + 256 + c] + Q[qrow + 256 + c];
            float xn = P[prow + 512 + c] + Q[qrow + 512 + c];
            float hr = s_hg[r * 768 + c], hz = s_hg[r * 768 + 256 + c], hn = s_hg[r * 768 + 512 + c];
            float rr = sigf(xr + hr), zz = sigf(xz + hz);
            float nn = tanhf(xn + rr * hn);
            float hnew = (1.f - zz) * nn + zz * s_h32[i];
            s_h32[i] = hnew;
            s_h16[i] = (half_t)hnew;
        }
        __syncthreads();

        // logit partials: 8 threads per (row, o) pair; intra-wave shuffle reduce
        {
            int g = tid >> 3, l8 = tid & 7;
            int r = g >> 1, o = g & 1;
            const float* wrow = W_hard + (size_t)o * 512 + hard_off;
            float s = 0.f;
            for (int c = l8; c < HD; c += 8) s += s_h32[r * HD + c] * wrow[c];
            s += __shfl_xor(s, 1, 32);
            s += __shfl_xor(s, 2, 32);
            s += __shfl_xor(s, 4, 32);
            if (l8 == 0) lout[((size_t)(row0 + r) * TT + t) * 2 + o] = s;
        }
        __syncthreads();
    }
}

// ---- Attention + hard gate + aggregation + value head (1 wave per row) ----
__global__ __launch_bounds__(256)
void attn_value_kernel(const float* __restrict__ h, const float* __restrict__ qb,
                       const float* __restrict__ kb, const float* __restrict__ lf,
                       const float* __restrict__ lr, const float* __restrict__ b_hard,
                       const float* __restrict__ gumbel, const float* __restrict__ Wv,
                       const float* __restrict__ bv, float* __restrict__ values) {
    int lane = threadIdx.x & 31, wave = threadIdx.x >> 5;
    int row = blockIdx.x * 8 + wave;
    int b = row >> 5, n = row & 31;
    int m = lane;
    int j = (m < n) ? m : m + 1;

    float sm = -3.4e38f;
    if (m < TT) {
        const float* qr = qb + (size_t)row * HD;
        const float* kr = kb + (size_t)(b * 32 + j) * HD;
        float s = 0.f;
        for (int c = 0; c < HD; ++c) s += qr[c] * kr[c];
        sm = s * (1.f / 16.f);   // 1/sqrt(256)
    }
    float mx = sm;
    for (int off = 16; off; off >>= 1) mx = fmaxf(mx, __shfl_xor(mx, off, 32));
    float ex = (m < TT) ? __expf(sm - mx) : 0.f;
    float den = ex;
    for (int off = 16; off; off >>= 1) den += __shfl_xor(den, off, 32);
    float softw = ex / den;

    float hard = 0.f;
    if (m < TT) {
        size_t li = ((size_t)row * TT + m) * 2;
        float y0 = lf[li + 0] + lr[li + 0] + b_hard[0] + gumbel[li + 0];
        float y1 = lf[li + 1] + lr[li + 1] + b_hard[1] + gumbel[li + 1];
        hard = (y1 > y0) ? 1.f : 0.f;   // straight-through == pure hard one-hot
    }
    float w = hard * softw;

    float acc[8] = {0.f, 0.f, 0.f, 0.f, 0.f, 0.f, 0.f, 0.f};
    for (int mm = 0; mm < TT; ++mm) {
        float wm = __shfl(w, mm, 32);
        int jj = (mm < n) ? mm : mm + 1;
        const float* hj = h + (size_t)(b * 32 + jj) * HD + lane * 8;
#pragma unroll
        for (int d = 0; d < 8; ++d) acc[d] += wm * hj[d];
    }
    float part = 0.f;
    const float* hrow = h + (size_t)row * HD + lane * 8;
#pragma unroll
    for (int d = 0; d < 8; ++d)
        part += hrow[d] * Wv[lane * 8 + d] + acc[d] * Wv[256 + lane * 8 + d];
    for (int off = 16; off; off >>= 1) part += __shfl_xor(part, off, 32);
    if (lane == 0) values[row] = part + bv[0];
}

// ---------------------------------------------------------------------------
extern "C" void kernel_launch(void* const* d_in, const int* in_sizes, int n_in,
                              void* d_out, int out_size, void* d_ws, size_t ws_size,
                              hipStream_t stream) {
    const float* cent_obs = (const float*)d_in[0];
    const float* lstm_h   = (const float*)d_in[1];
    const float* lstm_c   = (const float*)d_in[2];
    const float* masks    = (const float*)d_in[3];
    const float* gumbel   = (const float*)d_in[4];
    const float* W_enc    = (const float*)d_in[5];
    const float* b_enc    = (const float*)d_in[6];
    const float* lstm_Wi  = (const float*)d_in[7];
    const float* lstm_Wh  = (const float*)d_in[8];
    const float* lstm_bi  = (const float*)d_in[9];
    const float* lstm_bh  = (const float*)d_in[10];
    const float* gWi_f    = (const float*)d_in[11];
    const float* gWh_f    = (const float*)d_in[12];
    const float* gbi_f    = (const float*)d_in[13];
    const float* gbh_f    = (const float*)d_in[14];
    const float* gWi_b    = (const float*)d_in[15];
    const float* gWh_b    = (const float*)d_in[16];
    const float* gbi_b    = (const float*)d_in[17];
    const float* gbh_b    = (const float*)d_in[18];
    const float* W_hard   = (const float*)d_in[19];
    const float* b_hard   = (const float*)d_in[20];
    const float* Wq       = (const float*)d_in[21];
    const float* Wk       = (const float*)d_in[22];
    const float* Wv       = (const float*)d_in[23];
    const float* bv       = (const float*)d_in[24];

    float* values = (float*)d_out;
    float* out_h  = values + ROWS;
    float* out_c  = out_h + (size_t)ROWS * HD;

    char* ws = (char*)d_ws;
    size_t off = 0;
    auto walloc = [&](size_t bytes) { size_t o = off; off += (bytes + 255) & ~(size_t)255; return o; };

    half_t* hb16   = (half_t*)(ws + walloc((size_t)ROWS * HD * 2));
    half_t* we16   = (half_t*)(ws + walloc(256 * 512 * 2));
    half_t* wi16   = (half_t*)(ws + walloc(1024 * 256 * 2));
    half_t* wh16   = (half_t*)(ws + walloc(1024 * 256 * 2));
    half_t* wif16  = (half_t*)(ws + walloc(768 * 512 * 2));
    half_t* wib16  = (half_t*)(ws + walloc(768 * 512 * 2));
    half_t* whf16  = (half_t*)(ws + walloc(768 * 256 * 2));
    half_t* whb16  = (half_t*)(ws + walloc(768 * 256 * 2));
    half_t* wq16   = (half_t*)(ws + walloc(256 * 256 * 2));
    half_t* wk16   = (half_t*)(ws + walloc(256 * 256 * 2));
    float*  Pf     = (float*)(ws + walloc((size_t)ROWS * 768 * 4));
    float*  Qf     = (float*)(ws + walloc((size_t)ROWS * 768 * 4));
    float*  Pb     = (float*)(ws + walloc((size_t)ROWS * 768 * 4));
    float*  Qb     = (float*)(ws + walloc((size_t)ROWS * 768 * 4));
    float*  qbuf   = (float*)(ws + walloc((size_t)ROWS * HD * 4));
    float*  kbuf   = (float*)(ws + walloc((size_t)ROWS * HD * 4));
    float*  lf     = (float*)(ws + walloc((size_t)ROWS * TT * 2 * 4));
    float*  lr     = (float*)(ws + walloc((size_t)ROWS * TT * 2 * 4));

    auto cvt = [&](const float* s, half_t* d, int n) {
        cvt_f16_kernel<<<(n + 255) / 256, 256, 0, stream>>>(s, d, n);
    };
    cvt(W_enc,   we16, 256 * 512);
    cvt(lstm_Wi, wi16, 1024 * 256);
    cvt(lstm_Wh, wh16, 1024 * 256);
    cvt(gWi_f,   wif16, 768 * 512);
    cvt(gWi_b,   wib16, 768 * 512);
    cvt(gWh_f,   whf16, 768 * 256);
    cvt(gWh_b,   whb16, 768 * 256);
    cvt(Wq,      wq16, 256 * 256);
    cvt(Wk,      wk16, 256 * 256);

    enc_lstm_kernel<<<ROWS / 16, 256, 0, stream>>>(
        cent_obs, lstm_h, lstm_c, masks, we16, b_enc, wi16, wh16,
        lstm_bi, lstm_bh, out_h, out_c, hb16);

    // P/Q projections (GRU input decomposition) + attention q/k
    gemm16_kernel<<<dim3(ROWS / 16, 3), 256, 0, stream>>>(hb16, wif16, gbi_f, Pf, 512,   0, 768);
    gemm16_kernel<<<dim3(ROWS / 16, 3), 256, 0, stream>>>(hb16, wif16, nullptr, Qf, 512, 256, 768);
    gemm16_kernel<<<dim3(ROWS / 16, 3), 256, 0, stream>>>(hb16, wib16, gbi_b, Pb, 512,   0, 768);
    gemm16_kernel<<<dim3(ROWS / 16, 3), 256, 0, stream>>>(hb16, wib16, nullptr, Qb, 512, 256, 768);
    gemm16_kernel<<<dim3(ROWS / 16, 1), 256, 0, stream>>>(hb16, wq16, nullptr, qbuf, 256, 0, 256);
    gemm16_kernel<<<dim3(ROWS / 16, 1), 256, 0, stream>>>(hb16, wk16, nullptr, kbuf, 256, 0, 256);

    gru_kernel<<<ROWS / 16, 256, 0, stream>>>(Pf, Qf, whf16, gbh_f, W_hard,   0, lf, 0);
    gru_kernel<<<ROWS / 16, 256, 0, stream>>>(Pb, Qb, whb16, gbh_b, W_hard, 256, lr, 1);

    attn_value_kernel<<<ROWS / 8, 256, 0, stream>>>(
        out_h, qbuf, kbuf, lf, lr, b_hard, gumbel, Wv, bv, values);
}
